// SelfAttention_44676249813685
// MI455X (gfx1250) — matrix operand order repaired and checked by reading, not compiled, not run on hardware
//
#include <hip/hip_runtime.h>

// B=4, C=256, H=W=64, N=4096, CQK=32
// MI455X strategy: all GEMMs via v_wmma_f32_16x16x32_bf16 (wave32),
// x packed channel-last with a zero halo (no predication -> EXEC all-ones for WMMA),
// S^T materialized in bf16 (134MB, L2-resident on 192MB L2),
// softmax over the query axis = row-wise on S^T, 1/Z folded into the epilogue.

typedef __bf16 bf16;
typedef __attribute__((ext_vector_type(8)))  __bf16 v8bf;
typedef __attribute__((ext_vector_type(16))) __bf16 v16bf;
typedef __attribute__((ext_vector_type(8)))  float  v8f;

#define NB   4
#define CCH  256
#define HH   64
#define WW   64
#define NPIX 4096
#define DQK  32
#define HP   66   // padded H (halo of 1)
#define WP   66   // padded W (halo of 1)

// ---- WMMA fragment loaders (per CDNA5 ISA 16-bit layouts) -------------------
// A-matrix 16x32 (MxK): lane m=l&15; VGPR0-3 hold K = koff..koff+7 (koff=8*(l>=16)),
// VGPR4-7 hold K = 16+koff..16+koff+7.  -> two contiguous 16B loads per lane.
__device__ inline v16bf load_a_frag(const bf16* base, int row_stride, int lane) {
  const bf16* p = base + (size_t)(lane & 15) * row_stride + ((lane >> 4) << 3);
  v8bf lo = *(const v8bf*)(p);
  v8bf hi = *(const v8bf*)(p + 16);
  v16bf r;
#pragma unroll
  for (int i = 0; i < 8; ++i) { r[i] = lo[i]; r[i + 8] = hi[i]; }
  return r;
}

// B-matrix 32x16 (KxN) via B^T rows: lane n=l&15 reads K = koff..koff+7 and
// 16+koff..16+koff+7 (koff=8*(l>=16)) from row n of B^T (row-major, stride row_stride).
__device__ inline v16bf load_b_frag(const bf16* base, int row_stride, int lane) {
  const bf16* p = base + (size_t)(lane & 15) * row_stride + ((lane >> 4) << 3);
  v8bf lo = *(const v8bf*)(p);
  v8bf hi = *(const v8bf*)(p + 16);
  v16bf r;
#pragma unroll
  for (int i = 0; i < 8; ++i) { r[i] = lo[i]; r[i + 8] = hi[i]; }
  return r;
}

// ---- Pack x: [b][c][h][w] f32 -> channel-last padded [b][66][66][256] bf16 --
__global__ void pack_x_kernel(const float* __restrict__ x, bf16* __restrict__ xclp) {
  const int total = NB * HP * WP * CCH;
  int idx = blockIdx.x * blockDim.x + threadIdx.x;
  if (idx >= total) return;
  const int c   = idx & 255;
  int t         = idx >> 8;
  const int col = t % WP; t /= WP;
  const int ry  = t % HP; t /= HP;
  const int b   = t;
  bf16 v = (bf16)0.0f;
  if (ry >= 1 && ry <= HH && col >= 1 && col <= WW) {
    v = (bf16)x[((size_t)(b * CCH + c)) * NPIX + (ry - 1) * WW + (col - 1)];
  }
  xclp[idx] = v;
}

// ---- Pack weights: [cout][cin][3][3] f32 -> [cout 0..319][dydx 0..8][cin] bf16
__global__ void pack_w_kernel(const float* __restrict__ wq, const float* __restrict__ bq,
                              const float* __restrict__ wk, const float* __restrict__ bk,
                              const float* __restrict__ wv, const float* __restrict__ bv,
                              bf16* __restrict__ wpk, float* __restrict__ bpk) {
  const int NW = 320 * 9 * 256;
  int idx = blockIdx.x * blockDim.x + threadIdx.x;
  if (idx < NW) {
    const int ci   = idx & 255;
    const int dydx = (idx >> 8) % 9;
    const int co   = idx / (9 * 256);
    float v;
    if (co < 32)      v = wq[((co)      * 256 + ci) * 9 + dydx];
    else if (co < 64) v = wk[((co - 32) * 256 + ci) * 9 + dydx];
    else              v = wv[((co - 64) * 256 + ci) * 9 + dydx];
    wpk[idx] = (bf16)v;
  } else if (idx < NW + 320) {
    const int co = idx - NW;
    bpk[co] = (co < 32) ? bq[co] : (co < 64) ? bk[co - 32] : bv[co - 64];
  }
}

// ---- conv3x3 for Q,K (implicit GEMM, one 16x16 tile per wave) ---------------
// A = weights (16cout x 32cin), B = x-patch (32cin x 16pix). D lane: fixed pixel.
// ct 0-1 -> Q (d 0..31), ct 2-3 -> K. Store transposed [b][pix][32].
__global__ void conv_qk_kernel(const bf16* __restrict__ xclp, const bf16* __restrict__ wpk,
                               const float* __restrict__ bpk,
                               bf16* __restrict__ qcT, bf16* __restrict__ kcT) {
  const int lane = threadIdx.x & 31;
  int wid = blockIdx.x * (blockDim.x >> 5) + (threadIdx.x >> 5);
  const int ct = wid % 4;  wid /= 4;
  const int xt = wid % 4;  wid /= 4;
  const int y  = wid % 64; wid /= 64;
  const int b  = wid;
  const int x0  = xt * 16;
  const int row = lane & 15;
  const int hiq = lane >> 4;

  v8f acc{};
#pragma unroll
  for (int dydx = 0; dydx < 9; ++dydx) {
    const int dy0 = dydx / 3;   // 0..2 (padded row = y + dy0)
    const int dx0 = dydx % 3;   // 0..2 (padded col = x0 + dx0 + n)
    const bf16* xbase = xclp + ((size_t)((b * HP + (y + dy0)) * WP + (x0 + dx0))) * CCH;
    const bf16* wbase = wpk + ((size_t)(ct * 16) * 9 + dydx) * 256;
#pragma unroll
    for (int cc = 0; cc < 8; ++cc) {
      v16bf wf = load_a_frag(wbase + cc * 32, 9 * 256, lane);
      v16bf xf = load_b_frag(xbase + cc * 32, CCH, lane);
      acc = __builtin_amdgcn_wmma_f32_16x16x32_bf16(false, wf, false, xf,
                                                    (short)0, acc, false, false);
    }
  }
  // D: M = cout (j + 8*hiq), N = pixel (row)
  const int cb = ct * 16 + 8 * hiq;   // bias base (q:0..31, k:32..63)
  v8bf ov;
#pragma unroll
  for (int j = 0; j < 8; ++j) ov[j] = (bf16)(acc[j] + bpk[cb + j]);
  bf16* dst = (ct < 2 ? qcT : kcT);
  const int dcol = (ct & 1) * 16 + 8 * hiq;
  dst += ((size_t)b * NPIX + (y * WW + x0 + row)) * DQK + dcol;
  *(v8bf*)dst = ov;
}

// ---- conv3x3 for V ----------------------------------------------------------
// A = x-patch (16pix x 32cin), B = weights (32cin x 16cout). D lane: fixed cout.
// Store vc row-major [b][c][pix].
__global__ void conv_v_kernel(const bf16* __restrict__ xclp, const bf16* __restrict__ wpk,
                              const float* __restrict__ bpk, bf16* __restrict__ vc) {
  const int lane = threadIdx.x & 31;
  int wid = blockIdx.x * (blockDim.x >> 5) + (threadIdx.x >> 5);
  const int ct = wid % 16; wid /= 16;
  const int xt = wid % 4;  wid /= 4;
  const int y  = wid % 64; wid /= 64;
  const int b  = wid;
  const int x0  = xt * 16;
  const int row = lane & 15;
  const int hiq = lane >> 4;

  v8f acc{};
#pragma unroll
  for (int dydx = 0; dydx < 9; ++dydx) {
    const int dy0 = dydx / 3;
    const int dx0 = dydx % 3;
    const bf16* xbase = xclp + ((size_t)((b * HP + (y + dy0)) * WP + (x0 + dx0))) * CCH;
    const bf16* wbase = wpk + ((size_t)((4 + ct) * 16) * 9 + dydx) * 256;
#pragma unroll
    for (int cc = 0; cc < 8; ++cc) {
      v16bf xf = load_a_frag(xbase + cc * 32, CCH, lane);
      v16bf wf = load_b_frag(wbase + cc * 32, 9 * 256, lane);
      acc = __builtin_amdgcn_wmma_f32_16x16x32_bf16(false, xf, false, wf,
                                                    (short)0, acc, false, false);
    }
  }
  // D: M = pixel (j + 8*hiq), N = cout (row)
  const int cv = ct * 16 + row;
  const float bias = bpk[64 + cv];
  v8bf ov;
#pragma unroll
  for (int j = 0; j < 8; ++j) ov[j] = (bf16)(acc[j] + bias);
  bf16* dst = vc + ((size_t)(b * CCH + cv)) * NPIX + y * WW + x0 + 8 * hiq;
  *(v8bf*)dst = ov;
}

// ---- Scores: S = Qmat(NPIXx32) @ Kmat(32xNPIX); store S^T[k][q] bf16 --------
// One A fragment (q-tile) reused across 4 k-tiles per wave. Single K chunk (d=32).
__global__ void scores_kernel(const bf16* __restrict__ qcT, const bf16* __restrict__ kcT,
                              bf16* __restrict__ T) {
  const int lane = threadIdx.x & 31;
  int wid = blockIdx.x * (blockDim.x >> 5) + (threadIdx.x >> 5);
  const int ktg = wid % 64;  wid /= 64;
  const int qt  = wid % 256; wid /= 256;
  const int b   = wid;
  const int q0  = qt * 16;
  const int row = lane & 15;
  const int hiq = lane >> 4;

  const v16bf af = load_a_frag(qcT + ((size_t)b * NPIX + q0) * DQK, DQK, lane);
#pragma unroll
  for (int kt = 0; kt < 4; ++kt) {
    const int k0 = (ktg * 4 + kt) * 16;
    v16bf bfr = load_b_frag(kcT + ((size_t)b * NPIX + k0) * DQK, DQK, lane);
    v8f acc{};
    acc = __builtin_amdgcn_wmma_f32_16x16x32_bf16(false, af, false, bfr,
                                                  (short)0, acc, false, false);
    // D: M=q (j+8*hiq), N=k (row) -> store transposed row k, contiguous q
    v8bf ov;
#pragma unroll
    for (int j = 0; j < 8; ++j) ov[j] = (bf16)acc[j];
    bf16* dst = T + ((size_t)b * NPIX + (k0 + row)) * NPIX + q0 + 8 * hiq;
    *(v8bf*)dst = ov;
  }
}

// ---- Softmax over q (rows of S^T): in-place exp(S-m), save Z per row --------
__global__ void softmax_kernel(bf16* __restrict__ T, float* __restrict__ Z) {
  __shared__ float red[256];
  const size_t rowi = blockIdx.x;  // b*4096 + k
  bf16* p = T + rowi * NPIX;
  const int tid = threadIdx.x;
  float m = -1e30f;
  for (int i = tid; i < NPIX; i += 256) m = fmaxf(m, (float)p[i]);
  red[tid] = m; __syncthreads();
  for (int s = 128; s > 0; s >>= 1) { if (tid < s) red[tid] = fmaxf(red[tid], red[tid + s]); __syncthreads(); }
  m = red[0]; __syncthreads();
  float sum = 0.f;
  for (int i = tid; i < NPIX; i += 256) {
    float e = __expf((float)p[i] - m);
    sum += e;
    p[i] = (bf16)e;
  }
  red[tid] = sum; __syncthreads();
  for (int s = 128; s > 0; s >>= 1) { if (tid < s) red[tid] += red[tid + s]; __syncthreads(); }
  if (tid == 0) Z[rowi] = red[0];
}

// ---- O^T[k][c] = sum_q P^T[k][q] * V[c][q]; 2x2 register tiling -------------
// Per wave: k-tiles {k0, k0+16} x c-tiles {c0, c0+16}; 4 loads feed 4 WMMAs.
// Epilogue: out = x + g * O / Z[k], contiguous along k per lane.
__global__ void out_kernel(const bf16* __restrict__ PT, const bf16* __restrict__ vc,
                           const float* __restrict__ Z, const float* __restrict__ x,
                           const float* __restrict__ gamma, float* __restrict__ out) {
  const int lane = threadIdx.x & 31;
  int wid = blockIdx.x * (blockDim.x >> 5) + (threadIdx.x >> 5);
  const int cp = wid % 8;   wid /= 8;    // c-pair: 32 channels
  const int kp = wid % 128; wid /= 128;  // k-pair: 32 pixels
  const int b  = wid;
  const int k0 = kp * 32, c0 = cp * 32;
  const int row = lane & 15;
  const int hiq = lane >> 4;

  const bf16* A0 = PT + ((size_t)b * NPIX + k0) * NPIX;        // rows = k
  const bf16* A1 = A0 + (size_t)16 * NPIX;
  const bf16* B0 = vc + ((size_t)(b * CCH + c0)) * NPIX;       // B^T rows = c
  const bf16* B1 = B0 + (size_t)16 * NPIX;
  v8f acc00{}, acc01{}, acc10{}, acc11{};
#pragma unroll 2
  for (int qc = 0; qc < NPIX / 32; ++qc) {
    if (qc + 8 < NPIX / 32) {
      __builtin_prefetch((const void*)(A0 + (size_t)(qc + 8) * 32), 0, 0);
      __builtin_prefetch((const void*)(A1 + (size_t)(qc + 8) * 32), 0, 0);
      __builtin_prefetch((const void*)(B0 + (size_t)(qc + 8) * 32), 0, 0);
      __builtin_prefetch((const void*)(B1 + (size_t)(qc + 8) * 32), 0, 0);
    }
    v16bf a0 = load_a_frag(A0 + qc * 32, NPIX, lane);
    v16bf a1 = load_a_frag(A1 + qc * 32, NPIX, lane);
    v16bf b0 = load_b_frag(B0 + qc * 32, NPIX, lane);
    v16bf b1 = load_b_frag(B1 + qc * 32, NPIX, lane);
    acc00 = __builtin_amdgcn_wmma_f32_16x16x32_bf16(false, a0, false, b0, (short)0, acc00, false, false);
    acc01 = __builtin_amdgcn_wmma_f32_16x16x32_bf16(false, a0, false, b1, (short)0, acc01, false, false);
    acc10 = __builtin_amdgcn_wmma_f32_16x16x32_bf16(false, a1, false, b0, (short)0, acc10, false, false);
    acc11 = __builtin_amdgcn_wmma_f32_16x16x32_bf16(false, a1, false, b1, (short)0, acc11, false, false);
  }

  const float g = gamma[0];
#pragma unroll
  for (int tk = 0; tk < 2; ++tk) {
#pragma unroll
    for (int tc = 0; tc < 2; ++tc) {
      const v8f acc = (tk == 0) ? ((tc == 0) ? acc00 : acc01)
                                : ((tc == 0) ? acc10 : acc11);
      const int c  = c0 + tc * 16 + row;
      const int kb = k0 + tk * 16 + 8 * hiq;
      const float* zp = Z + (size_t)b * NPIX + kb;
      const float* xp = x + ((size_t)(b * CCH + c)) * NPIX + kb;
      float*       op = out + ((size_t)(b * CCH + c)) * NPIX + kb;
      const float4 xv0 = *(const float4*)(xp);
      const float4 xv1 = *(const float4*)(xp + 4);
      const float4 zv0 = *(const float4*)(zp);
      const float4 zv1 = *(const float4*)(zp + 4);
      float4 o0, o1;
      o0.x = xv0.x + g * acc[0] / zv0.x;
      o0.y = xv0.y + g * acc[1] / zv0.y;
      o0.z = xv0.z + g * acc[2] / zv0.z;
      o0.w = xv0.w + g * acc[3] / zv0.w;
      o1.x = xv1.x + g * acc[4] / zv1.x;
      o1.y = xv1.y + g * acc[5] / zv1.y;
      o1.z = xv1.z + g * acc[6] / zv1.z;
      o1.w = xv1.w + g * acc[7] / zv1.w;
      *(float4*)(op)     = o0;
      *(float4*)(op + 4) = o1;
    }
  }
}

// ---- Host-side orchestration ------------------------------------------------
extern "C" void kernel_launch(void* const* d_in, const int* in_sizes, int n_in,
                              void* d_out, int out_size, void* d_ws, size_t ws_size,
                              hipStream_t stream) {
  const float* x     = (const float*)d_in[0];
  const float* wq    = (const float*)d_in[1];
  const float* bq    = (const float*)d_in[2];
  const float* wk    = (const float*)d_in[3];
  const float* bk    = (const float*)d_in[4];
  const float* wv    = (const float*)d_in[5];
  const float* bv    = (const float*)d_in[6];
  const float* gamma = (const float*)d_in[7];
  float* out = (float*)d_out;

  char* ws = (char*)d_ws;
  size_t off = 0;
  auto carve = [&](size_t bytes) -> void* {
    void* p = ws + off;
    off += (bytes + 255) & ~(size_t)255;
    return p;
  };
  bf16*  xclp = (bf16*)carve((size_t)NB * HP * WP * CCH * 2);  // 8.9 MB (zero halo)
  bf16*  wpk  = (bf16*)carve((size_t)320 * 9 * 256 * 2);       // 1.5 MB
  float* bpk  = (float*)carve((size_t)320 * 4);
  bf16*  qcT  = (bf16*)carve((size_t)NB * NPIX * DQK * 2);     // 1 MB
  bf16*  kcT  = (bf16*)carve((size_t)NB * NPIX * DQK * 2);     // 1 MB
  bf16*  vc   = (bf16*)carve((size_t)NB * CCH * NPIX * 2);     // 8.4 MB
  float* Zb   = (float*)carve((size_t)NB * NPIX * 4);
  bf16*  T    = (bf16*)carve((size_t)NB * NPIX * NPIX * 2);    // 134 MB (L2-resident)

  pack_x_kernel<<<(NB * HP * WP * CCH + 255) / 256, 256, 0, stream>>>(x, xclp);
  pack_w_kernel<<<(320 * 9 * 256 + 320 + 255) / 256, 256, 0, stream>>>(
      wq, bq, wk, bk, wv, bv, wpk, bpk);
  // conv QK: waves = 4b * 64y * 4xt * 4ct = 4096; 8 waves/block
  conv_qk_kernel<<<4096 / 8, 256, 0, stream>>>(xclp, wpk, bpk, qcT, kcT);
  // conv V: waves = 4b * 64y * 4xt * 16ct = 16384
  conv_v_kernel<<<16384 / 8, 256, 0, stream>>>(xclp, wpk, bpk, vc);
  // scores: waves = 4b * 256qt * 64ktg = 65536
  scores_kernel<<<65536 / 8, 256, 0, stream>>>(qcT, kcT, T);
  softmax_kernel<<<NB * NPIX, 256, 0, stream>>>(T, Zb);
  // out: waves = 4b * 128kp * 8cp = 4096
  out_kernel<<<4096 / 8, 256, 0, stream>>>(T, vc, Zb, x, gamma, out);
}
